// imagetoclass_42417097015420
// MI455X (gfx1250) — matrix operand order repaired and checked by reading, not compiled, not run on hardware
//
#include <hip/hip_runtime.h>
#include <hip/hip_bf16.h>

// Problem constants (from reference)
#define C_CLASS 5
#define SHOTS   5
#define DIMS    64
#define HW      256          // 16*16 spatial locations
#define NQ      75           // number of query images
#define M_PER_C (SHOTS*HW)   // 1280 support descriptors per class
#define KTOP    3
#define K2TOP   64

typedef _Float16 v16h __attribute__((ext_vector_type(16)));
typedef _Float16 v8h  __attribute__((ext_vector_type(8)));
typedef float    v8f  __attribute__((ext_vector_type(8)));

// ---------------------------------------------------------------------------
// Stage 1: L2-normalize each 64-dim local descriptor and pack to f16,
// descriptor-major (contiguous K) so WMMA fragments are contiguous loads.
//   Sh[c][m][k]  c=class, m = shot*256+pix, k=0..63
//   Qh[b][q][k]  q = pixel index
// ---------------------------------------------------------------------------
__global__ __launch_bounds__(256) void norm_pack_kernel(
    const float* __restrict__ sup, const float* __restrict__ qry,
    _Float16* __restrict__ Sh, _Float16* __restrict__ Qh) {
  int gid = blockIdx.x * 256 + threadIdx.x;
  const int NSUP = C_CLASS * SHOTS * HW;   // 6400
  if (gid < NSUP) {
    int n = gid / HW, pix = gid % HW;
    const float* v = sup + (size_t)n * DIMS * HW + pix;
    float tmp[DIMS];
    float ss = 0.f;
#pragma unroll
    for (int d = 0; d < DIMS; ++d) { float x = v[(size_t)d * HW]; tmp[d] = x; ss += x * x; }
    float rn = __frsqrt_rn(ss);
    int c = n / SHOTS, shot = n % SHOTS;
    _Float16* o = Sh + (((size_t)c * M_PER_C) + (size_t)shot * HW + pix) * DIMS;
#pragma unroll
    for (int d = 0; d < DIMS; ++d) o[d] = (_Float16)(tmp[d] * rn);
  } else {
    int qi = gid - NSUP;
    if (qi >= NQ * HW) return;
    int b = qi / HW, pix = qi % HW;
    const float* v = qry + (size_t)b * DIMS * HW + pix;
    float tmp[DIMS];
    float ss = 0.f;
#pragma unroll
    for (int d = 0; d < DIMS; ++d) { float x = v[(size_t)d * HW]; tmp[d] = x; ss += x * x; }
    float rn = __frsqrt_rn(ss);
    _Float16* o = Qh + ((size_t)b * HW + pix) * DIMS;
#pragma unroll
    for (int d = 0; d < DIMS; ++d) o[d] = (_Float16)(tmp[d] * rn);
  }
}

// ---------------------------------------------------------------------------
// A-fragment loader: 16-bit A-matrix 16x32, contiguous k order (same as B).
// Lane L (L<16): row = rowbase+L, halves 0..15 = K kbase+0..15.
// Lane L+16: same row, K offset +16.  Both 16B-aligned loads.
// ---------------------------------------------------------------------------
__device__ __forceinline__ v16h load_a_frag(const _Float16* __restrict__ Qb,
                                            int rowbase, int l16, int hi, int kbase) {
  const _Float16* p = Qb + (size_t)(rowbase + l16) * DIMS + kbase + hi * 16;
  v8h lo = *(const v8h*)p;
  v8h hh = *(const v8h*)(p + 8);
  v16h a;
#pragma unroll
  for (int t = 0; t < 8; ++t) { a[t] = lo[t]; a[t + 8] = hh[t]; }
  return a;
}

// ---------------------------------------------------------------------------
// Stage 2: fused sim-GEMM + running top-3 over the 1280 support descriptors.
// One block per (c,b).  8 waves; wave w owns query rows [32w, 32w+32).
// For each 16-wide m-tile: 4x v_wmma_f32_16x16x32_f16 (K=64 as 2 chained
// WMMAs), tile -> LDS, each thread folds its row's 16 sims into a branchless
// sorted top-3 (keeps EXEC uniform around the WMMAs).
// ---------------------------------------------------------------------------
__global__ __launch_bounds__(256) void sim_top3_kernel(
    const _Float16* __restrict__ Sh, const _Float16* __restrict__ Qh,
    float* __restrict__ v1) {
  __shared__ float sim[HW * 16];   // 16 KB

  int bid = blockIdx.x;            // 0..374
  int c = bid / NQ, b = bid % NQ;
  int tid = threadIdx.x;
  int lane = tid & 31;
  int wave = tid >> 5;
  int l16 = lane & 15;
  int hi  = lane >> 4;             // 0: lanes 0-15, 1: lanes 16-31
  int rowbase = wave * 32;

  const _Float16* Qb = Qh + (size_t)b * HW * DIMS;
  // A fragments: reused across all 80 m-tiles (register resident)
  v16h a00 = load_a_frag(Qb, rowbase,      l16, hi, 0);
  v16h a01 = load_a_frag(Qb, rowbase,      l16, hi, 32);
  v16h a10 = load_a_frag(Qb, rowbase + 16, l16, hi, 0);
  v16h a11 = load_a_frag(Qb, rowbase + 16, l16, hi, 32);

  float t0 = -3.4e38f, t1 = -3.4e38f, t2 = -3.4e38f;
  const _Float16* Sc = Sh + (size_t)c * M_PER_C * DIMS;

  for (int mt = 0; mt < M_PER_C / 16; ++mt) {
    // B-fragment: 32x16 f16 layout — lane n<16: col = mt*16+n, K 0..15;
    // lane n+16: same col, K 16..31.  Contiguous 32B per lane.
    const _Float16* sb = Sc + (size_t)(mt * 16 + l16) * DIMS + hi * 16;
    v16h b0 = *(const v16h*)sb;          // K = 0..31 slice
    v16h b1 = *(const v16h*)(sb + 32);   // K = 32..63 slice

    v8f acc0 = {};
    acc0 = __builtin_amdgcn_wmma_f32_16x16x32_f16(false, a00, false, b0, (short)0, acc0, false, false);
    acc0 = __builtin_amdgcn_wmma_f32_16x16x32_f16(false, a01, false, b1, (short)0, acc0, false, false);
    v8f acc1 = {};
    acc1 = __builtin_amdgcn_wmma_f32_16x16x32_f16(false, a10, false, b0, (short)0, acc1, false, false);
    acc1 = __builtin_amdgcn_wmma_f32_16x16x32_f16(false, a11, false, b1, (short)0, acc1, false, false);

    // C/D layout: lane holds col l16; VGPR r holds row rowbase + hi*8 + r
#pragma unroll
    for (int r = 0; r < 8; ++r) {
      sim[(rowbase +      hi * 8 + r) * 16 + l16] = acc0[r];
      sim[(rowbase + 16 + hi * 8 + r) * 16 + l16] = acc1[r];
    }
    __syncthreads();

    // Thread t owns query row t: fold 16 new sims into sorted top-3
#pragma unroll
    for (int j = 0; j < 16; ++j) {
      float v  = sim[tid * 16 + j];
      float m0 = fmaxf(t0, v), r0 = fminf(t0, v);
      float m1 = fmaxf(t1, r0), r1 = fminf(t1, r0);
      float m2 = fmaxf(t2, r1);
      t0 = m0; t1 = m1; t2 = m2;
    }
    __syncthreads();   // readers done before next tile overwrites LDS
  }

  // v1 layout: [(c*NQ+b)*3 + j][256]  (row-major over query locations)
  size_t base = ((size_t)c * NQ + b) * KTOP * HW;
  v1[base            + tid] = t0;
  v1[base +     HW   + tid] = t1;
  v1[base + 2 * HW   + tid] = t2;
}

// ---------------------------------------------------------------------------
// Stage 3: top-64 (sorted descending) of each 256-row -> output row of 64.
// Descending bitonic sort in LDS, one block per (c,b,j) row.
// ---------------------------------------------------------------------------
__global__ __launch_bounds__(256) void topk2_kernel(
    const float* __restrict__ v1, float* __restrict__ out) {
  __shared__ float s[HW];
  int r = blockIdx.x;        // 0..1124  = (c*NQ+b)*3 + j
  int t = threadIdx.x;
  s[t] = v1[(size_t)r * HW + t];
  __syncthreads();
#pragma unroll 1
  for (int k = 2; k <= HW; k <<= 1) {
#pragma unroll 1
    for (int j = k >> 1; j > 0; j >>= 1) {
      int ixj = t ^ j;
      if (ixj > t) {
        float a = s[t], bb = s[ixj];
        bool desc = ((t & k) == 0);
        if (desc ? (a < bb) : (a > bb)) { s[t] = bb; s[ixj] = a; }
      }
      __syncthreads();
    }
  }
  if (t < K2TOP) out[(size_t)r * K2TOP + t] = s[t];
}

// ---------------------------------------------------------------------------
extern "C" void kernel_launch(void* const* d_in, const int* in_sizes, int n_in,
                              void* d_out, int out_size, void* d_ws, size_t ws_size,
                              hipStream_t stream) {
  (void)in_sizes; (void)n_in; (void)out_size; (void)ws_size;
  const float* sup = (const float*)d_in[0];   // [25,64,16,16] f32
  const float* qry = (const float*)d_in[1];   // [75,64,16,16] f32
  // d_in[2..4] = task_index, k(=3), k2(=64): fixed by the problem, hardcoded.

  char* ws = (char*)d_ws;
  const size_t SH_BYTES = (size_t)C_CLASS * M_PER_C * DIMS * sizeof(_Float16); // 819200
  const size_t QH_BYTES = (size_t)NQ * HW * DIMS * sizeof(_Float16);           // 2457600
  _Float16* Sh = (_Float16*)ws;
  _Float16* Qh = (_Float16*)(ws + SH_BYTES);
  float*    v1 = (float*)(ws + SH_BYTES + QH_BYTES);                           // 1152000 B
  float*    out = (float*)d_out;                                               // [375,1,3,64]

  const int n_desc = C_CLASS * SHOTS * HW + NQ * HW;   // 25600
  norm_pack_kernel<<<(n_desc + 255) / 256, 256, 0, stream>>>(sup, qry, Sh, Qh);
  sim_top3_kernel<<<C_CLASS * NQ, 256, 0, stream>>>(Sh, Qh, v1);
  topk2_kernel<<<C_CLASS * NQ * KTOP, 256, 0, stream>>>(v1, out);
}
